// MixtureOfExpertsHead_86320252715284
// MI455X (gfx1250) — matrix operand order repaired and checked by reading, not compiled, not run on hardware
//
#include <hip/hip_runtime.h>

// ---------------- MoE head, MI455X (gfx1250) ----------------
// bf16 WMMA for the fused [16384,4096] x [4096, 9*2048] GEMM
// (gate + 8 dense experts), layer-2 (O=1, E=8) folded into epilogue.
// Weights transposed+converted to bf16 once (151 MB -> L2-resident),
// then streamed into LDS with the Tensor Data Mover (double-buffered).
// A tiles (fp32 x -> bf16 via native cvt) are double-buffered too, so
// both staging paths overlap the WMMA compute interval.

#define Bdim 16384
#define Hdim 4096
#define H2d  2048
#define Ed   8
#define BLK_M 64     // rows per workgroup (4 M-tiles)
#define NBC  128     // columns per n-block (8 N-tiles)
#define KT   32      // K per WMMA step
#define KSTEPS (Hdim / KT)

typedef __attribute__((ext_vector_type(16))) __bf16 v16bf;
typedef __attribute__((ext_vector_type(8)))  __bf16 v8bf;
typedef __attribute__((ext_vector_type(8)))  float  v8f;
typedef unsigned int u32x4 __attribute__((ext_vector_type(4)));
typedef int          i32x4 __attribute__((ext_vector_type(4)));
typedef int          i32x8 __attribute__((ext_vector_type(8)));

union Frag16 { v16bf v; uint4 q[2]; };

__device__ inline unsigned short f2bf(float f) {
  unsigned int u = __float_as_uint(f);
  unsigned int r = u + 0x7FFFu + ((u >> 16) & 1u);   // round-nearest-even
  return (unsigned short)(r >> 16);
}

// Issue a TDM load of a 32(k) x 128(n) bf16 tile -> LDS [n][64B] layout.
__device__ inline void tdm_load_b_tile(const unsigned short* gsrc,
                                       unsigned int lds_off) {
  unsigned long long ga = (unsigned long long)gsrc;
  u32x4 g0;
  g0[0] = 1u;                                      // count=1, user mode
  g0[1] = lds_off;                                 // LDS byte address
  g0[2] = (unsigned int)(ga & 0xffffffffull);      // global_addr[31:0]
  g0[3] = (unsigned int)((ga >> 32) & 0x01ffffffu) // global_addr[56:32]
          | (2u << 30);                            // type=2 ("image")
  i32x8 g1;
  g1[0] = (1 << 16);          // data_size=1 -> 2 bytes/elem
  g1[1] = (KT << 16);         // tensor_dim0 = 32 (low 16 bits)
  g1[2] = (NBC << 16);        // tensor_dim0 hi=0 | tensor_dim1 = 128
  g1[3] = (KT << 16);         // tensor_dim1 hi=0 | tile_dim0 = 32
  g1[4] = NBC;                // tile_dim1 = 128, tile_dim2 = 0
  g1[5] = Hdim;               // tensor_dim0_stride = 4096 (lo 32)
  g1[6] = 0;                  // stride hi | tensor_dim1_stride lo
  g1[7] = 0;
  i32x4 g2;
  g2[0] = 1;                  // tensor_dim2 = 1 (benign)
  g2[1] = 1;                  // tensor_dim3 = 1
  g2[2] = 0;                  // tensor_dim2_stride lo
  g2[3] = 0;                  // stride hi | tile_dim3 = 0
  i32x4 g3;
  g3[0] = 0;                  // tensor_dim3_stride lo
  g3[1] = (1 << 16);          // stride hi | tensor_dim4 = 1
  g3[2] = 0;                  // tile_dim4 = 0
  g3[3] = 0;
#if __clang_major__ >= 23
  i32x8 g4 = {0, 0, 0, 0, 0, 0, 0, 0};
  __builtin_amdgcn_tensor_load_to_lds(g0, g1, g2, g3, g4, 0);
#else
  __builtin_amdgcn_tensor_load_to_lds(g0, g1, g2, g3, 0);
#endif
}

// ---- Kernel 1: transpose + convert [Wg1 | We1] -> Wt[9][H2][H] bf16 ----
__global__ __launch_bounds__(256)
void transpose_convert_w(const float* __restrict__ Wg1,
                         const float* __restrict__ We1,
                         unsigned short* __restrict__ Wt) {
  __shared__ float tile[32][33];
  const int g  = blockIdx.z;           // 0 = gate, 1..8 = experts
  const int d0 = blockIdx.x * 32;      // H2 index
  const int h0 = blockIdx.y * 32;      // H index
  const float* src = (g == 0) ? Wg1 : (We1 + (size_t)(g - 1) * Hdim * H2d);
  const int tx = threadIdx.x, ty = threadIdx.y;  // (32, 8)
#pragma unroll
  for (int i = 0; i < 4; ++i) {
    int h = h0 + ty + i * 8;
    tile[ty + i * 8][tx] = src[(size_t)h * H2d + (d0 + tx)];
  }
  __syncthreads();
#pragma unroll
  for (int i = 0; i < 4; ++i) {
    int d = d0 + ty + i * 8;
    Wt[((size_t)g * H2d + d) * Hdim + (h0 + tx)] = f2bf(tile[tx][ty + i * 8]);
  }
}

// ---- Kernel 2: fused gate + all-expert MLP + top-2 combine ----
__global__ __launch_bounds__(256)
void moe_head_kernel(const float* __restrict__ x,
                     const unsigned short* __restrict__ Wt,   // [9][H2][H] bf16
                     const float* __restrict__ be1,           // [E][H2]
                     const float* __restrict__ We2,           // [E][H2] (O=1)
                     const float* __restrict__ be2,           // [E]
                     const float* __restrict__ bg1,           // [H2]
                     const float* __restrict__ Wg2,           // [H2][E]
                     const float* __restrict__ bg2,           // [E]
                     float* __restrict__ out) {
  __shared__ __align__(16) unsigned short ldsA[2][BLK_M * KT];    // 8 KB dbl
  __shared__ __align__(16) unsigned short ldsB[2][NBC * KT];      // 16 KB dbl
  __shared__ float ldsLogit[4][16][Ed];                           // 2 KB
  __shared__ float ldsSum[4][16][Ed];                             // 2 KB

  const int tid  = threadIdx.x;
  const int wave = tid >> 5;
  const int lane = tid & 31;
  const int b0   = blockIdx.x * BLK_M;

  // init accumulators with layer-2 biases
  for (int i = tid; i < 4 * 16 * Ed; i += 256) {
    (&ldsLogit[0][0][0])[i] = bg2[i & 7];
    (&ldsSum[0][0][0])[i]   = be2[i & 7];
  }
  __syncthreads();

  // wave -> 2 M-tiles x 2 N-tiles (LDS-op : WMMA ratio = 2.0)
  const int mt0  = (wave >> 2) * 2;   // 0 or 2
  const int nt0  = (wave & 3) * 2;    // 0,2,4,6
  const int nloc = lane & 15;
  const int hi   = lane >> 4;

  const unsigned int ldsBoff[2] = {
      (unsigned int)(unsigned long long)(void*)&ldsB[0][0],
      (unsigned int)(unsigned long long)(void*)&ldsB[1][0]};

  // A staging: 64 rows x 32 k, fp32 -> bf16 with native packed cvt
  const int sm = tid >> 2, sch = tid & 3;
  const float* xrow = x + (size_t)(b0 + sm) * Hdim + sch * 8;

  for (int nb = 0; nb < 144; ++nb) {
    const int g  = nb >> 4;                   // weight group
    const int c0 = (nb & 15) * NBC;           // column base inside group
    const unsigned short* wbase = Wt + ((size_t)g * H2d + c0) * Hdim;

    v8f acc[2][2];
#pragma unroll
    for (int mi = 0; mi < 2; ++mi)
#pragma unroll
      for (int ni = 0; ni < 2; ++ni) acc[mi][ni] = (v8f)(0.0f);

    // prologue: stage kt=0 (A via VALU cvt, B via TDM)
    if (wave == 0) tdm_load_b_tile(wbase, ldsBoff[0]);
    {
      v8f f = *(const v8f*)xrow;
      *(v8bf*)&ldsA[0][sm * KT + sch * 8] = __builtin_convertvector(f, v8bf);
    }

    for (int kt = 0; kt < KSTEPS; ++kt) {
      const int cur = kt & 1, nxt = (kt + 1) & 1;
      __syncthreads();   // compute(kt-1) + stage(kt) complete
      if (wave == 0) {
        if (kt + 1 < KSTEPS) {
          tdm_load_b_tile(wbase + (size_t)(kt + 1) * KT, ldsBoff[nxt]);
          __builtin_amdgcn_s_wait_tensorcnt(1);   // TDM(kt) complete
        } else {
          __builtin_amdgcn_s_wait_tensorcnt(0);
        }
      }
      __syncthreads();   // B(kt) handed over to all waves

      // ---- overlap: stage A(kt+1) while computing kt ----
      if (kt + 1 < KSTEPS) {
        v8f f = *(const v8f*)(xrow + (kt + 1) * KT);
        *(v8bf*)&ldsA[nxt][sm * KT + sch * 8] = __builtin_convertvector(f, v8bf);
        __builtin_prefetch(xrow + (kt + 2) * KT, 0, 1);
      }

      const unsigned short* abuf = &ldsA[cur][0];
      const unsigned short* bbuf = &ldsB[cur][0];
      Frag16 a[2], b[2];
#pragma unroll
      for (int mi = 0; mi < 2; ++mi) {
        const uint4* arow = (const uint4*)&abuf[((mt0 + mi) * 16 + nloc) * KT];
        a[mi].q[0] = arow[hi];
        a[mi].q[1] = arow[2 + hi];
      }
#pragma unroll
      for (int ni = 0; ni < 2; ++ni) {
        const uint4* brow = (const uint4*)&bbuf[((nt0 + ni) * 16 + nloc) * KT];
        b[ni].q[0] = brow[hi];
        b[ni].q[1] = brow[2 + hi];
      }
#pragma unroll
      for (int mi = 0; mi < 2; ++mi)
#pragma unroll
        for (int ni = 0; ni < 2; ++ni)
          acc[mi][ni] = __builtin_amdgcn_wmma_f32_16x16x32_bf16(
              false, a[mi].v, false, b[ni].v, (short)0, acc[mi][ni], false, false);
    }

    // ---- epilogue: bias + ReLU, then dot with layer-2 weights ----
    int gcol[2];
    const float* biasp = (g == 0) ? bg1 : (be1 + (size_t)(g - 1) * H2d);
#pragma unroll
    for (int ni = 0; ni < 2; ++ni) {
      gcol[ni] = c0 + (nt0 + ni) * 16 + nloc;
      float bias = biasp[gcol[ni]];
#pragma unroll
      for (int mi = 0; mi < 2; ++mi)
#pragma unroll
        for (int v = 0; v < 8; ++v) {
          float t = acc[mi][ni][v] + bias;
          acc[mi][ni][v] = t > 0.0f ? t : 0.0f;
        }
    }
    if (g == 0) {
      // gate: logits[m][e] += sum_cols relu * Wg2[col][e]
      for (int e = 0; e < Ed; ++e) {
        float part[2][8];
#pragma unroll
        for (int mi = 0; mi < 2; ++mi)
#pragma unroll
          for (int v = 0; v < 8; ++v) part[mi][v] = 0.0f;
#pragma unroll
        for (int ni = 0; ni < 2; ++ni) {
          float w = Wg2[(size_t)gcol[ni] * Ed + e];
#pragma unroll
          for (int mi = 0; mi < 2; ++mi)
#pragma unroll
            for (int v = 0; v < 8; ++v) part[mi][v] += acc[mi][ni][v] * w;
        }
#pragma unroll
        for (int off = 8; off >= 1; off >>= 1)
#pragma unroll
          for (int mi = 0; mi < 2; ++mi)
#pragma unroll
            for (int v = 0; v < 8; ++v)
              part[mi][v] += __shfl_xor(part[mi][v], off, 32);
        if (nloc == 0)
#pragma unroll
          for (int mi = 0; mi < 2; ++mi)
#pragma unroll
            for (int v = 0; v < 8; ++v)
              atomicAdd(&ldsLogit[mt0 + mi][v + 8 * hi][e], part[mi][v]);
      }
    } else {
      // expert e: scalar_out[m] += sum_cols relu * We2[e][col]
      const int e = g - 1;
      const float* w2 = We2 + (size_t)e * H2d;
      float part[2][8];
#pragma unroll
      for (int mi = 0; mi < 2; ++mi)
#pragma unroll
        for (int v = 0; v < 8; ++v) part[mi][v] = 0.0f;
#pragma unroll
      for (int ni = 0; ni < 2; ++ni) {
        float w = w2[gcol[ni]];
#pragma unroll
        for (int mi = 0; mi < 2; ++mi)
#pragma unroll
          for (int v = 0; v < 8; ++v) part[mi][v] += acc[mi][ni][v] * w;
      }
#pragma unroll
      for (int off = 8; off >= 1; off >>= 1)
#pragma unroll
        for (int mi = 0; mi < 2; ++mi)
#pragma unroll
          for (int v = 0; v < 8; ++v)
            part[mi][v] += __shfl_xor(part[mi][v], off, 32);
      if (nloc == 0)
#pragma unroll
        for (int mi = 0; mi < 2; ++mi)
#pragma unroll
          for (int v = 0; v < 8; ++v)
            atomicAdd(&ldsSum[mt0 + mi][v + 8 * hi][e], part[mi][v]);
    }
  }

  __syncthreads();
  // ---- finale: softmax -> top-2 -> renormalize == sigmoid(l1-l2) blend ----
  if (tid < BLK_M) {
    const int mt = tid >> 4, r = tid & 15;
    float lg[Ed], sm2[Ed];
#pragma unroll
    for (int e = 0; e < Ed; ++e) { lg[e] = ldsLogit[mt][r][e]; sm2[e] = ldsSum[mt][r][e]; }
    int i1 = 0;
#pragma unroll
    for (int e = 1; e < Ed; ++e) if (lg[e] > lg[i1]) i1 = e;
    int i2 = (i1 == 0) ? 1 : 0;
#pragma unroll
    for (int e = 0; e < Ed; ++e) if (e != i1 && lg[e] > lg[i2]) i2 = e;
    float w = 1.0f / (1.0f + __expf(lg[i2] - lg[i1]));
    out[b0 + tid] = w * sm2[i1] + (1.0f - w) * sm2[i2];
  }
}

extern "C" void kernel_launch(void* const* d_in, const int* in_sizes, int n_in,
                              void* d_out, int out_size, void* d_ws, size_t ws_size,
                              hipStream_t stream) {
  const float* x   = (const float*)d_in[0];
  const float* We1 = (const float*)d_in[1];
  const float* be1 = (const float*)d_in[2];
  const float* We2 = (const float*)d_in[3];
  const float* be2 = (const float*)d_in[4];
  const float* Wg1 = (const float*)d_in[5];
  const float* bg1 = (const float*)d_in[6];
  const float* Wg2 = (const float*)d_in[7];
  const float* bg2 = (const float*)d_in[8];

  // workspace: Wt bf16 [9][2048][4096] = 150.9 MB (L2-resident on MI455X)
  unsigned short* Wt = (unsigned short*)d_ws;

  dim3 tb(32, 8);
  dim3 tg(H2d / 32, Hdim / 32, 9);
  transpose_convert_w<<<tg, tb, 0, stream>>>(Wg1, We1, Wt);

  moe_head_kernel<<<Bdim / BLK_M, 256, 0, stream>>>(
      x, Wt, be1, We2, be2, bg1, Wg2, bg2, (float*)d_out);
}
